// KCN_56642028699847
// MI455X (gfx1250) — matrix operand order repaired and checked by reading, not compiled, not run on hardware
//
#include <hip/hip_runtime.h>

typedef __attribute__((ext_vector_type(16))) _Float16 v16h;
typedef __attribute__((ext_vector_type(8)))  float    v8f;

#define KN   25     // nodes per subgraph
#define PM   32     // padded node dim (2 WMMA M-tiles)
#define FIN  128    // input features
#define FH   256    // hidden features
#define DEG  10
#define EPB  250    // edges per block
#define NTHREADS 256

// LDS strides (elements) — padded to spread banks, kept even for u32 pair loads
#define XS   132    // sX16 [32][128] f16
#define W0S  132    // sW0T [64][128] f16 (chunk of W0^T)
#define ATS  34     // sA16 [32][32] f16
#define XWTS 34     // sXWT [256][32] f16  (XW transposed: B operand of GEMM2)
#define H1S  264    // sH1  [32][256] f16

struct alignas(16) Smem {
    float a[PM * PM];          // adjacency Â (f32, atomic build), row-major [tgt][src]
    float deg[32];
    float dinv[32];
    _Float16 a16[PM * ATS];    // f16 copy of Â (A operand of GEMM2)
    union {
        struct {
            _Float16 x16[PM * XS];    // X block f16 (A operand of GEMM1)
            _Float16 w0t[64 * W0S];   // W0^T chunk f16 (B operand of GEMM1)
        } g1;
        _Float16 h1[PM * H1S];        // relu(Â @ XW) — aliases dead X/W0 staging
    } u;
    _Float16 xwt[FH * XWTS];   // (X@W0)^T f16
    float v[NTHREADS];
    float red[NTHREADS];
};

__global__ __launch_bounds__(NTHREADS)
void gcn_block_kernel(const float* __restrict__ x,
                      const float* __restrict__ ew,
                      const float* __restrict__ W0,
                      const float* __restrict__ W1,
                      const float* __restrict__ Wlin,
                      const int*   __restrict__ tgt,
                      float*       __restrict__ out)
{
    __shared__ Smem sm;
    const int b    = blockIdx.x;
    const int t    = threadIdx.x;
    const int lane = t & 31;          // wave32
    const int wave = t >> 5;          // 8 waves
    const int l15  = lane & 15;
    const int hi   = lane >> 4;

    // ---------------- zero init ----------------
    for (int i = t; i < PM * PM; i += NTHREADS) sm.a[i] = 0.f;
    if (t < 32) { sm.deg[t] = 0.f; sm.dinv[t] = 0.f; }
    for (int i = t; i < PM * XS; i += NTHREADS) sm.u.g1.x16[i] = (_Float16)0.f;
    __syncthreads();

    // ---------------- adjacency build ----------------
    float w = 0.f; int ls = 0, lt = 0;
    if (t < EPB) {
        const int e = b * EPB + t;
        ls = t / DEG;                 // src is local t/DEG by construction
        lt = tgt[e] - b * KN;
        w  = ew[e];
        atomicAdd(&sm.deg[lt], w);
    }
    __syncthreads();
    if (t < KN) {
        const float d = sm.deg[t];
        sm.dinv[t] = d > 0.f ? (1.0f / sqrtf(d)) : 0.f;
    }
    __syncthreads();
    if (t < EPB)
        atomicAdd(&sm.a[lt * PM + ls], sm.dinv[ls] * w * sm.dinv[lt]);

    // ---------------- stage X block as f16 ----------------
    for (int i = t; i < KN * FIN; i += NTHREADS) {
        const int m = i / FIN, k = i % FIN;
        sm.u.g1.x16[m * XS + k] = (_Float16)x[(b * KN + m) * FIN + k];
    }
    __syncthreads();

    // f16 copy of Â
    for (int i = t; i < PM * PM; i += NTHREADS) {
        const int m = i / PM, k = i % PM;
        sm.a16[m * ATS + k] = (_Float16)sm.a[i];
    }
    __syncthreads();

    // ---------------- GEMM1: XW = X(32x128) @ W0(128x256), f16 WMMA ----------------
    // 4 chunks of 64 hidden cols; per chunk each wave owns one 16x16 tile.
    for (int c = 0; c < 4; ++c) {
        for (int i = t; i < 64 * FIN; i += NTHREADS) {
            const int nn = i % 64, k = i / 64;          // coalesced over nn
            sm.u.g1.w0t[nn * W0S + k] = (_Float16)W0[k * FH + c * 64 + nn];
        }
        __syncthreads();

        const int mtile = wave & 1;
        const int ntl   = wave >> 1;                    // 0..3 within chunk
        v8f acc = {};
        const unsigned* ap = (const unsigned*)sm.u.g1.x16;
        const unsigned* bp = (const unsigned*)sm.u.g1.w0t;
        const int arow = (mtile * 16 + l15) * XS;
        const int brow = (ntl * 16 + l15) * W0S;
        #pragma unroll
        for (int kk = 0; kk < 4; ++kk) {
            union { v16h v; unsigned q[8]; } A, B;
            #pragma unroll
            for (int j = 0; j < 8; ++j) {               // A: f16 16x32 layout
                const int kb = (j < 4 ? 2 * j : 16 + 2 * (j - 4)) + hi * 8 + kk * 32;
                A.q[j] = ap[(arow + kb) >> 1];
            }
            #pragma unroll
            for (int j = 0; j < 8; ++j) {               // B: f16 32x16 layout (from B^T rows)
                const int kb = (j < 4 ? 2 * j : 16 + 2 * (j - 4)) + hi * 8 + kk * 32;
                B.q[j] = bp[(brow + kb) >> 1];
            }
            acc = __builtin_amdgcn_wmma_f32_16x16x32_f16(
                      false, A.v, false, B.v, (short)0, acc, false, false);
        }
        // store tile transposed: element (m,n) -> xwt[n][m]
        const int nrow  = (c * 64 + ntl * 16 + l15) * XWTS;
        const int mbase = mtile * 16 + hi * 8;
        #pragma unroll
        for (int r = 0; r < 8; ++r)
            sm.xwt[nrow + mbase + r] = (_Float16)acc[r];
        __syncthreads();
    }

    // ---------------- GEMM2: H1 = relu( Â(32x32) @ XW(32x256) ) ----------------
    #pragma unroll
    for (int tt = 0; tt < 4; ++tt) {
        const int tile  = wave + tt * 8;                // 0..31
        const int mtile = tile & 1;
        const int ntile = tile >> 1;                    // 0..15
        union { v16h v; unsigned q[8]; } A, B;
        const unsigned* ap = (const unsigned*)sm.a16;
        const unsigned* bp = (const unsigned*)sm.xwt;
        const int arow = (mtile * 16 + l15) * ATS;
        const int brow = (ntile * 16 + l15) * XWTS;
        #pragma unroll
        for (int j = 0; j < 8; ++j) {
            const int kb = (j < 4 ? 2 * j : 16 + 2 * (j - 4)) + hi * 8;
            A.q[j] = ap[(arow + kb) >> 1];
        }
        #pragma unroll
        for (int j = 0; j < 8; ++j) {
            const int kb = (j < 4 ? 2 * j : 16 + 2 * (j - 4)) + hi * 8;
            B.q[j] = bp[(brow + kb) >> 1];
        }
        v8f acc = {};
        acc = __builtin_amdgcn_wmma_f32_16x16x32_f16(
                  false, A.v, false, B.v, (short)0, acc, false, false);
        const int n     = ntile * 16 + l15;
        const int mbase = mtile * 16 + hi * 8;
        #pragma unroll
        for (int r = 0; r < 8; ++r) {
            const float rv = acc[r] > 0.f ? acc[r] : 0.f;
            sm.u.h1[(mbase + r) * H1S + n] = (_Float16)rv;   // h1 aliases dead X/W0 staging
        }
    }
    __syncthreads();

    // ---------------- second layer, row-0 only ----------------
    // out_b = relu( (Â[0,:] @ h1) @ W1 ) · Wlin
    {
        float acc = 0.f;
        #pragma unroll
        for (int m = 0; m < KN; ++m)
            acc += sm.a[m] * (float)sm.u.h1[m * H1S + t];    // Â row 0
        sm.v[t] = acc;
    }
    __syncthreads();
    float uacc = 0.f;
    for (int i = 0; i < FH; ++i)
        uacc += sm.v[i] * W1[i * FH + t];                    // coalesced over t
    sm.red[t] = (uacc > 0.f ? uacc : 0.f) * Wlin[t];
    __syncthreads();
    for (int s = 128; s > 0; s >>= 1) {
        if (t < s) sm.red[t] += sm.red[t + s];
        __syncthreads();
    }
    if (t == 0) out[b] = sm.red[0];
}

extern "C" void kernel_launch(void* const* d_in, const int* in_sizes, int n_in,
                              void* d_out, int out_size, void* d_ws, size_t ws_size,
                              hipStream_t stream) {
    (void)n_in; (void)d_ws; (void)ws_size;
    const float* x    = (const float*)d_in[0];
    const float* ew   = (const float*)d_in[1];
    const float* W0   = (const float*)d_in[2];
    const float* W1   = (const float*)d_in[3];
    const float* Wlin = (const float*)d_in[4];
    const int*   ei   = (const int*)d_in[5];   // [2, E]: row0 = src, row1 = tgt
    const int    E    = in_sizes[1];
    const int*   tgt  = ei + E;
    float* out = (float*)d_out;
    const int B = out_size;                    // 2048 subgraphs
    gcn_block_kernel<<<B, NTHREADS, 0, stream>>>(x, ew, W0, W1, Wlin, tgt, out);
}
